// Informer_26852135534693
// MI455X (gfx1250) — matrix operand order repaired and checked by reading, not compiled, not run on hardware
//
#include <hip/hip_runtime.h>
#include <math.h>

// ---------------------------------------------------------------------------
// Informer forward for MI455X (gfx1250, wave32, WMMA).
// All dense GEMMs go through v_wmma_f32_16x16x32_f16 (f16 in / f32 acc).
// LDS tiles are stored in WMMA fragment order so operand loads are b128s.
// ProbSparse attention bookkeeping (sampling, top-u, scatter) stays VALU.
// ---------------------------------------------------------------------------

typedef __attribute__((ext_vector_type(16))) _Float16 v16h;
typedef __attribute__((ext_vector_type(8)))  _Float16 v8h;
typedef __attribute__((ext_vector_type(8)))  float    v8f;

#define DM  512   // d_model
#define NH  8     // heads
#define DH  64    // head dim
#define DFF 2048
#define BATCH 8

// ======================= generic WMMA GEMM =================================
// C = act(alpha * A(MxK) x B + bias [+ Cprev]), B either [N,K] (bNK=1, with
// k-stride ldbK) or [K,N] (bNK=0).  Per-z base offsets:
//   base = (z/zdiv)*outer + (z%zdiv)*inner   (elements)
struct GemmP {
  const float* A; const float* Bw; const float* bias; float* C;
  int M, N, K, lda, ldb, ldc, ldbK;
  int zdiv;
  long long aO, aI, bO, bI, cO, cI;
  float alpha;
  int act;    // 0=none 1=gelu(exact)
  int bNK;    // 1: Bw[n*ldb + k*ldbK]   0: Bw[k*ldb + n]
  int accum;  // 1: C += result
};

// Block tile 64M x 64N, 8 wave32s (4 along M x 2 along N), each wave owns two
// 16x16 C tiles sharing one A fragment -> 2 v_wmma per wave per K-step of 32.
__global__ __launch_bounds__(256) void gemm_f16_wmma(GemmP p) {
  // LDS in WMMA fragment order: [16-row group][lane][16 halves] (32B per lane)
  __shared__ __align__(32) _Float16 AsF[4][32][16];
  __shared__ __align__(32) _Float16 BsF[4][32][16];

  const int z = blockIdx.z;
  const long long aBase = (long long)(z / p.zdiv) * p.aO + (long long)(z % p.zdiv) * p.aI;
  const long long bBase = (long long)(z / p.zdiv) * p.bO + (long long)(z % p.zdiv) * p.bI;
  const long long cBase = (long long)(z / p.zdiv) * p.cO + (long long)(z % p.zdiv) * p.cI;

  const int tid  = threadIdx.x;
  const int lane = tid & 31;
  const int wave = tid >> 5;
  const int wm   = wave & 3;          // 4 waves along M
  const int wn   = wave >> 2;         // 2 waves along N (x2 subtiles each)
  const int tileM = blockIdx.y * 64;
  const int tileN = blockIdx.x * 64;
  const int half  = lane & 15;

  const bool fullM = (tileM + 64 <= p.M);
  const bool fullN = (tileN + 64 <= p.N);
  const bool alignedA = ((p.lda & 3) == 0) && ((aBase & 3) == 0) &&
                        ((((size_t)p.A) & 15) == 0);
  const bool alignedB = p.bNK && (p.ldbK == 1) && ((p.ldb & 3) == 0) &&
                        ((bBase & 3) == 0) && ((((size_t)p.Bw) & 15) == 0);

  // A staging: thread -> (row am, 8-wide k-chunk ac). Fragment mapping:
  //   ac=0: lane lo,e0..7 | ac=1: lane hi,e0..7 | ac=2: lane lo,e8..15 | ac=3: hi,e8..15
  const int am = tid >> 2, ac = tid & 3;
  const int alane = (am & 15) + ((ac & 1) << 4);
  const int ae0   = (ac & 2) ? 8 : 0;
  // B staging: thread -> (col bn, 8-wide k-chunk bc). Fragment mapping:
  //   bc=0: lane lo,e0..7 | bc=1: lane hi,e0..7 | bc=2: lane lo,e8..15 | bc=3: hi,e8..15
  const int bn = tid >> 2, bc = tid & 3;
  const int blane = (bn & 15) + ((bc & 1) << 4);
  const int be0   = (bc & 2) ? 8 : 0;

  v8f acc0 = {0.f,0.f,0.f,0.f,0.f,0.f,0.f,0.f};
  v8f acc1 = {0.f,0.f,0.f,0.f,0.f,0.f,0.f,0.f};

  for (int k0 = 0; k0 < p.K; k0 += 32) {
    const bool fullK = (k0 + 32 <= p.K);

    // ---------------- stage A (64x32) ----------------
    if (fullM && fullK) {
      const float* src = p.A + aBase + (long long)(tileM + am) * p.lda + k0 + ac * 8;
      v8h hv;
      if (alignedA) {
        const float4 f0 = ((const float4*)src)[0];
        const float4 f1 = ((const float4*)src)[1];
        hv[0]=(_Float16)f0.x; hv[1]=(_Float16)f0.y; hv[2]=(_Float16)f0.z; hv[3]=(_Float16)f0.w;
        hv[4]=(_Float16)f1.x; hv[5]=(_Float16)f1.y; hv[6]=(_Float16)f1.z; hv[7]=(_Float16)f1.w;
      } else {
#pragma unroll
        for (int j = 0; j < 8; ++j) hv[j] = (_Float16)src[j];
      }
      *(v8h*)&AsF[am >> 4][alane][ae0] = hv;
    } else {
      for (int i = tid; i < 64 * 32; i += 256) {
        int m = i >> 5, kk = i & 31;
        int mg = tileM + m, kg = k0 + kk;
        float v = 0.f;
        if (mg < p.M && kg < p.K) v = p.A[aBase + (long long)mg * p.lda + kg];
        int la = (m & 15) + (((kk >> 3) & 1) << 4);
        int e  = (kk & 7) + ((kk >= 16) ? 8 : 0);
        AsF[m >> 4][la][e] = (_Float16)v;
      }
    }

    // ---------------- stage B (32k x 64n) ----------------
    if (fullN && fullK) {
      v8h hv;
      if (alignedB) {
        const float* src = p.Bw + bBase + (long long)(tileN + bn) * p.ldb + k0 + bc * 8;
        const float4 f0 = ((const float4*)src)[0];
        const float4 f1 = ((const float4*)src)[1];
        hv[0]=(_Float16)f0.x; hv[1]=(_Float16)f0.y; hv[2]=(_Float16)f0.z; hv[3]=(_Float16)f0.w;
        hv[4]=(_Float16)f1.x; hv[5]=(_Float16)f1.y; hv[6]=(_Float16)f1.z; hv[7]=(_Float16)f1.w;
      } else if (p.bNK) {
        const float* src = p.Bw + bBase + (long long)(tileN + bn) * p.ldb
                         + (long long)(k0 + bc * 8) * p.ldbK;
#pragma unroll
        for (int j = 0; j < 8; ++j) hv[j] = (_Float16)src[(long long)j * p.ldbK];
      } else {
        const float* src = p.Bw + bBase + (long long)(k0 + bc * 8) * p.ldb + tileN + bn;
#pragma unroll
        for (int j = 0; j < 8; ++j) hv[j] = (_Float16)src[(long long)j * p.ldb];
      }
      *(v8h*)&BsF[bn >> 4][blane][be0] = hv;
    } else {
      for (int i = tid; i < 32 * 64; i += 256) {
        int kk = i >> 6, n = i & 63;
        int ng = tileN + n, kg = k0 + kk;
        float v = 0.f;
        if (ng < p.N && kg < p.K)
          v = p.bNK ? p.Bw[bBase + (long long)ng * p.ldb + (long long)kg * p.ldbK]
                    : p.Bw[bBase + (long long)kg * p.ldb + ng];
        BsF[n >> 4][(n & 15) + (((kk >> 3) & 1) << 4)][(kk & 7) + ((kk >= 16) ? 8 : 0)] = (_Float16)v;
      }
    }

    // prefetch next K-slab of A (global_prefetch_b8)
    if (k0 + 32 < p.K && fullM)
      __builtin_prefetch(&p.A[aBase + (long long)(tileM + am) * p.lda + (k0 + 32)], 0, 0);

    __syncthreads();

    // ---------------- compute: 2 WMMAs sharing the A fragment ----------------
    v16h af  = *(const v16h*)&AsF[wm][lane][0];
    v16h bf0 = *(const v16h*)&BsF[wn * 2 + 0][lane][0];
    v16h bf1 = *(const v16h*)&BsF[wn * 2 + 1][lane][0];
    acc0 = __builtin_amdgcn_wmma_f32_16x16x32_f16(false, af, false, bf0,
                                                  (short)0, acc0, false, false);
    acc1 = __builtin_amdgcn_wmma_f32_16x16x32_f16(false, af, false, bf1,
                                                  (short)0, acc1, false, false);
    __syncthreads();
  }

  // ---- epilogue: C layout VGPR r -> (M = r / r+8, N = lane%16) ----
#pragma unroll
  for (int j = 0; j < 2; ++j) {
    const v8f& acc = j ? acc1 : acc0;
    int col = tileN + (wn * 2 + j) * 16 + half;
    if (col >= p.N) continue;
#pragma unroll
    for (int r = 0; r < 8; ++r) {
      int row = tileM + wm * 16 + ((lane < 16) ? r : r + 8);
      if (row < p.M) {
        long long ci = cBase + (long long)row * p.ldc + col;
        float v = acc[r] * p.alpha;
        if (p.accum) v += p.C[ci];
        if (p.bias)  v += p.bias[col];
        if (p.act == 1) v = 0.5f * v * (1.0f + erff(v * 0.70710678118654752f));
        p.C[ci] = v;
      }
    }
  }
}

// ======================= elementwise / reduction kernels ===================

__global__ void embed_kernel(const float* __restrict__ x, const float* __restrict__ xm,
                             const float* __restrict__ cw, const float* __restrict__ tw,
                             const float* __restrict__ tb, const float* __restrict__ pe,
                             float* __restrict__ out, int L, int Cin) {
  long long i = (long long)blockIdx.x * blockDim.x + threadIdx.x;
  long long total = (long long)BATCH * L * DM;
  if (i >= total) return;
  int c = (int)(i % DM); long long r = i / DM;
  int l = (int)(r % L);  int b = (int)(r / L);
  float v = 0.f;
  for (int t = 0; t < 3; ++t) {                 // circular conv, pad=1
    int ls = (l + t - 1 + L) % L;
    const float* xr = x + ((long long)b * L + ls) * Cin;
    const float* wr = cw + (long long)c * Cin * 3 + t;
    for (int ci = 0; ci < Cin; ++ci) v += xr[ci] * wr[ci * 3];
  }
  const float* mr = xm + ((long long)b * L + l) * 4;
  const float* wr = tw + (long long)c * 4;
  for (int j = 0; j < 4; ++j) v += mr[j] * wr[j];
  out[i] = v + tb[c] + pe[(long long)l * DM + c];
}

__global__ __launch_bounds__(256) void ln_kernel(const float* __restrict__ a,
                                                 const float* __restrict__ b2,
                                                 const float* __restrict__ g,
                                                 const float* __restrict__ bb,
                                                 float* __restrict__ out, int rows) {
  int wave = threadIdx.x >> 5, lane = threadIdx.x & 31;
  int row = blockIdx.x * 8 + wave;
  if (row >= rows) return;
  const float* pa = a + (long long)row * DM;
  const float* pb = b2 ? b2 + (long long)row * DM : nullptr;
  float s = 0.f, s2 = 0.f;
  for (int j = lane; j < DM; j += 32) {
    float v = pa[j] + (pb ? pb[j] : 0.f);
    s += v; s2 += v * v;
  }
  for (int o = 16; o > 0; o >>= 1) { s += __shfl_xor(s, o, 32); s2 += __shfl_xor(s2, o, 32); }
  float mean = s * (1.f / DM);
  float var  = s2 * (1.f / DM) - mean * mean;
  float inv  = rsqrtf(var + 1e-5f);
  float* po = out + (long long)row * DM;
  for (int j = lane; j < DM; j += 32) {
    float v = pa[j] + (pb ? pb[j] : 0.f);
    po[j] = (v - mean) * inv * g[j] + bb[j];
  }
}

__global__ __launch_bounds__(256) void softmax_rows(float* __restrict__ s, int rows, int len) {
  int wave = threadIdx.x >> 5, lane = threadIdx.x & 31;
  int row = blockIdx.x * 8 + wave;
  if (row >= rows) return;
  float* p = s + (long long)row * len;
  float mx = -INFINITY;
  for (int j = lane; j < len; j += 32) mx = fmaxf(mx, p[j]);
  for (int o = 16; o > 0; o >>= 1) mx = fmaxf(mx, __shfl_xor(mx, o, 32));
  float sum = 0.f;
  for (int j = lane; j < len; j += 32) sum += expf(p[j] - mx);
  for (int o = 16; o > 0; o >>= 1) sum += __shfl_xor(sum, o, 32);
  float inv = 1.f / sum;
  for (int j = lane; j < len; j += 32) p[j] = expf(p[j] - mx) * inv;
}

// ---- ProbSparse attention machinery ----
__global__ void sample_m_kernel(const float* __restrict__ q, const float* __restrict__ k,
                                float* __restrict__ M, int L, int LK, int U, unsigned seed) {
  long long t = (long long)blockIdx.x * blockDim.x + threadIdx.x;
  long long total = (long long)BATCH * NH * L;
  if (t >= total) return;
  int l = (int)(t % L); int h = (int)((t / L) % NH); int b = (int)(t / ((long long)L * NH));
  const float* qp = q + ((long long)b * L + l) * DM + h * DH;
  float mx = -INFINITY, sm = 0.f;
  for (int j = 0; j < U; ++j) {
    unsigned x = ((unsigned)l * 1103515245u) ^ ((unsigned)j * 2654435761u) ^ seed;
    x ^= x >> 16; x *= 0x85EBCA6Bu; x ^= x >> 13; x *= 0xC2B2AE35u; x ^= x >> 16;
    int idx = (int)(x % (unsigned)LK);
    const float* kp = k + ((long long)b * LK + idx) * DM + h * DH;
    float d = 0.f;
    for (int e = 0; e < DH; ++e) d += qp[e] * kp[e];
    mx = fmaxf(mx, d); sm += d;
  }
  M[((long long)b * NH + h) * L + l] = mx - sm / (float)LK;
}

__global__ __launch_bounds__(32) void topk_kernel(const float* __restrict__ M,
                                                  int* __restrict__ idxTop, int L, int u) {
  __shared__ float cache[2048];
  int row = blockIdx.x, lane = threadIdx.x;
  const float* mp = M + (long long)row * L;
  for (int j = lane; j < L; j += 32) cache[j] = mp[j];
  __syncthreads();
  for (int t = 0; t < u; ++t) {
    float bv = -INFINITY; int bi = 0x7fffffff;
    for (int j = lane; j < L; j += 32) {
      float v = cache[j];
      if (v > bv || (v == bv && j < bi)) { bv = v; bi = j; }
    }
    for (int o = 16; o > 0; o >>= 1) {
      float ov = __shfl_xor(bv, o, 32); int oi = __shfl_xor(bi, o, 32);
      if (ov > bv || (ov == bv && oi < bi)) { bv = ov; bi = oi; }
    }
    if (lane == 0) { idxTop[(long long)row * u + t] = bi; cache[bi] = -INFINITY; }
    __syncthreads();
  }
}

__global__ void scores_red_kernel(const float* __restrict__ q, const float* __restrict__ k,
                                  const int* __restrict__ idxTop, float* __restrict__ sc,
                                  int L, int LK, int u, int maskFlag) {
  long long t = (long long)blockIdx.x * blockDim.x + threadIdx.x;
  long long total = (long long)BATCH * NH * u * LK;
  if (t >= total) return;
  int s = (int)(t % LK); long long r = t / LK;
  int i = (int)(r % u); r /= u;
  int h = (int)(r % NH); int b = (int)(r / NH);
  int l = idxTop[((long long)b * NH + h) * u + i];
  float v;
  if (maskFlag && s > l) v = -INFINITY;
  else {
    const float* qp = q + ((long long)b * L + l) * DM + h * DH;
    const float* kp = k + ((long long)b * LK + s) * DM + h * DH;
    float d = 0.f;
    for (int e = 0; e < DH; ++e) d += qp[e] * kp[e];
    v = d * 0.125f;   // 1/sqrt(64)
  }
  sc[(((long long)b * NH + h) * u + i) * LK + s] = v;
}

__global__ void vmean_kernel(const float* __restrict__ v, float* __restrict__ vm, int S) {
  int t = blockIdx.x * blockDim.x + threadIdx.x;
  if (t >= BATCH * NH * DH) return;
  int d = t % DH; int h = (t / DH) % NH; int b = t / (DH * NH);
  float s = 0.f;
  for (int j = 0; j < S; ++j) s += v[((long long)b * S + j) * DM + h * DH + d];
  vm[((long long)b * NH + h) * DH + d] = s / (float)S;
}

__global__ void ctx_bcast_kernel(const float* __restrict__ vm, float* __restrict__ ctx, int L) {
  long long t = (long long)blockIdx.x * blockDim.x + threadIdx.x;
  long long total = (long long)BATCH * NH * L * DH;
  if (t >= total) return;
  int d = (int)(t % DH); long long r = t / DH;
  r /= L;                       // skip l
  int h = (int)(r % NH); int b = (int)(r / NH);
  ctx[t] = vm[((long long)b * NH + h) * DH + d];
}

__global__ void ctx_cumsum_kernel(const float* __restrict__ v, float* __restrict__ ctx, int L) {
  int t = blockIdx.x * blockDim.x + threadIdx.x;
  if (t >= BATCH * NH * DH) return;
  int d = t % DH; int h = (t / DH) % NH; int b = t / (DH * NH);
  float acc = 0.f;
  for (int l = 0; l < L; ++l) {
    acc += v[((long long)b * L + l) * DM + h * DH + d];
    ctx[(((long long)b * NH + h) * L + l) * DH + d] = acc;
  }
}

__global__ void ctx_scatter_kernel(const float* __restrict__ pr, const float* __restrict__ v,
                                   const int* __restrict__ idxTop, float* __restrict__ ctx,
                                   int L, int LK, int u) {
  long long t = (long long)blockIdx.x * blockDim.x + threadIdx.x;
  long long total = (long long)BATCH * NH * u * DH;
  if (t >= total) return;
  int d = (int)(t % DH); long long r = t / DH;
  int i = (int)(r % u); r /= u;
  int h = (int)(r % NH); int b = (int)(r / NH);
  int l = idxTop[((long long)b * NH + h) * u + i];
  const float* pp = pr + (((long long)b * NH + h) * u + i) * LK;
  float s = 0.f;
  for (int j = 0; j < LK; ++j) s += pp[j] * v[((long long)b * LK + j) * DM + h * DH + d];
  ctx[(((long long)b * NH + h) * L + l) * DH + d] = s;
}

__global__ void unmix_kernel(const float* __restrict__ ctx, float* __restrict__ out, int L) {
  long long t = (long long)blockIdx.x * blockDim.x + threadIdx.x;
  long long total = (long long)BATCH * L * DM;
  if (t >= total) return;
  int c = (int)(t % DM); long long r = t / DM;
  int l = (int)(r % L); int b = (int)(r / L);
  int h = c / DH, d = c % DH;
  out[t] = ctx[(((long long)b * NH + h) * L + l) * DH + d];
}

__global__ void wrappad_kernel(const float* __restrict__ x, float* __restrict__ xp,
                               int L, int pad) {
  long long t = (long long)blockIdx.x * blockDim.x + threadIdx.x;
  int Lp = L + 2 * pad;
  long long total = (long long)BATCH * Lp * DM;
  if (t >= total) return;
  int c = (int)(t % DM); long long r = t / DM;
  int i = (int)(r % Lp); int b = (int)(r / Lp);
  int ls = (i - pad + L) % L;
  xp[t] = x[((long long)b * L + ls) * DM + c];
}

__global__ void bnelu_kernel(float* __restrict__ y, const float* __restrict__ bm,
                             const float* __restrict__ bv, const float* __restrict__ bg,
                             const float* __restrict__ bb, long long n) {
  long long t = (long long)blockIdx.x * blockDim.x + threadIdx.x;
  if (t >= n) return;
  int c = (int)(t % DM);
  float v = (y[t] - bm[c]) * rsqrtf(bv[c] + 1e-5f) * bg[c] + bb[c];
  y[t] = (v > 0.f) ? v : expm1f(v);
}

__global__ void maxpool3_kernel(const float* __restrict__ y, float* __restrict__ out,
                                int Lc, int Lp) {
  long long t = (long long)blockIdx.x * blockDim.x + threadIdx.x;
  long long total = (long long)BATCH * Lp * DM;
  if (t >= total) return;
  int c = (int)(t % DM); long long r = t / DM;
  int lp = (int)(r % Lp); int b = (int)(r / Lp);
  int l0 = 2 * lp - 1;
  float m = -INFINITY;
  for (int tt = 0; tt < 3; ++tt) {
    int l = l0 + tt;
    if (l >= 0 && l < Lc) m = fmaxf(m, y[((long long)b * Lc + l) * DM + c]);
  }
  out[t] = m;
}

__global__ void slice_out_kernel(const float* __restrict__ p, float* __restrict__ out,
                                 int L, int pred, int C) {
  long long t = (long long)blockIdx.x * blockDim.x + threadIdx.x;
  long long total = (long long)BATCH * pred * C;
  if (t >= total) return;
  int c = (int)(t % C); long long r = t / C;
  int l = (int)(r % pred); int b = (int)(r / pred);
  out[t] = p[((long long)b * L + (L - pred + l)) * C + c];
}

// ======================= host orchestration ================================

static inline dim3 grid1(long long n) { return dim3((unsigned)((n + 255) / 256)); }

static void launch_gemm(hipStream_t st, const float* A, const float* Bw, const float* bias,
                        float* C, int M, int N, int K, int lda, int ldb, int ldc,
                        float alpha, int act, int bNK, int accum, int nz, int zdiv,
                        long long aO, long long aI, long long bO, long long bI,
                        long long cO, long long cI, int ldbK) {
  GemmP p;
  p.A = A; p.Bw = Bw; p.bias = bias; p.C = C;
  p.M = M; p.N = N; p.K = K; p.lda = lda; p.ldb = ldb; p.ldc = ldc; p.ldbK = ldbK;
  p.zdiv = (zdiv < 1) ? 1 : zdiv;
  p.aO = aO; p.aI = aI; p.bO = bO; p.bI = bI; p.cO = cO; p.cI = cI;
  p.alpha = alpha; p.act = act; p.bNK = bNK; p.accum = accum;
  dim3 g((N + 63) / 64, (M + 63) / 64, nz);
  hipLaunchKernelGGL(gemm_f16_wmma, g, dim3(256), 0, st, p);
}

struct AttnW { const float *wq,*bq,*wk,*bk,*wv,*bv,*wo,*bo; };
struct EncL  { AttnW a; const float *w1,*b1,*w2,*b2,*ln1g,*ln1b,*ln2g,*ln2b; };
struct ConvD { const float *w,*b,*bng,*bnb,*bnm,*bnv; };
struct DecL  { AttnW self, cross;
               const float *w1,*b1,*w2,*b2,*ln1g,*ln1b,*ln2g,*ln2b,*ln3g,*ln3b; };

extern "C" void kernel_launch(void* const* d_in, const int* in_sizes, int n_in,
                              void* d_out, int out_size, void* d_ws, size_t ws_size,
                              hipStream_t stream) {
  (void)in_sizes; (void)n_in; (void)out_size; (void)ws_size;
  // ---- unpack inputs in setup_inputs() flattening order ----
  int pi = 0;
  auto nf = [&]() { return (const float*)d_in[pi++]; };
  const float* x_enc = nf(); const float* x_mark_enc = nf();
  const float* x_dec = nf(); const float* x_mark_dec = nf();
  const float *eCw = nf(), *eTw = nf(), *eTb = nf(), *ePe = nf();
  const float *dCw = nf(), *dTw = nf(), *dTb = nf(), *dPe = nf();
  auto rdAttn = [&]() { AttnW a; a.wq=nf();a.bq=nf();a.wk=nf();a.bk=nf();
                        a.wv=nf();a.bv=nf();a.wo=nf();a.bo=nf(); return a; };
  EncL enc[3];
  for (int i = 0; i < 3; ++i) {
    enc[i].a = rdAttn();
    enc[i].w1=nf(); enc[i].b1=nf(); enc[i].w2=nf(); enc[i].b2=nf();
    enc[i].ln1g=nf(); enc[i].ln1b=nf(); enc[i].ln2g=nf(); enc[i].ln2b=nf();
  }
  ConvD cv[2];
  for (int i = 0; i < 2; ++i) { cv[i].w=nf(); cv[i].b=nf(); cv[i].bng=nf();
                                cv[i].bnb=nf(); cv[i].bnm=nf(); cv[i].bnv=nf(); }
  const float *encNg = nf(), *encNb = nf();
  DecL dec[2];
  for (int i = 0; i < 2; ++i) {
    dec[i].self = rdAttn(); dec[i].cross = rdAttn();
    dec[i].w1=nf(); dec[i].b1=nf(); dec[i].w2=nf(); dec[i].b2=nf();
    dec[i].ln1g=nf(); dec[i].ln1b=nf(); dec[i].ln2g=nf(); dec[i].ln2b=nf();
    dec[i].ln3g=nf(); dec[i].ln3b=nf();
  }
  const float *decNg = nf(), *decNb = nf();
  const float *projW = nf(), *projB = nf();

  // ---- workspace bump allocator ----
  size_t off = 0;
  auto allocF = [&](long long elems) {
    float* p = (float*)((char*)d_ws + off);
    off += ((size_t)elems * 4 + 255) & ~(size_t)255;
    return p;
  };
  const long long ACT = (long long)BATCH * 2048 * DM;
  float* bufA = allocF(ACT);
  float* bufB = allocF(ACT);
  float* bufC = allocF(ACT);
  float* bufQ = allocF(ACT);
  float* bufK = allocF(ACT);
  float* bufV = allocF(ACT);
  float* encOut = allocF(ACT);
  float* big  = allocF((long long)BATCH * NH * 1024 * 520 > (long long)BATCH * 2048 * DFF
                         ? (long long)BATCH * NH * 1024 * 520
                         : (long long)BATCH * 2048 * DFF);
  float* ctxBuf = allocF((long long)BATCH * NH * 2048 * DH);
  float* scRed  = allocF((long long)BATCH * NH * 40 * 2048);
  float* Mbuf   = allocF((long long)BATCH * NH * 2048);
  float* vmeanB = allocF((long long)BATCH * NH * DH);
  int*   idxTop = (int*)allocF((long long)BATCH * NH * 40);
  float* convXp = allocF((long long)BATCH * 2052 * DM);
  float* convY  = allocF((long long)BATCH * 2050 * DM);
  float* projY  = allocF((long long)BATCH * 1024 * 7);

  auto lin = [&](const float* A, const float* W, const float* bias, float* C,
                 int M, int N, int K, int act) {
    launch_gemm(stream, A, W, bias, C, M, N, K, K, K, N,
                1.f, act, 1, 0, 1, 1, 0, 0, 0, 0, 0, 0, 1);
  };
  auto ln = [&](const float* a, const float* b2, const float* g, const float* bb,
                float* out, int rows) {
    hipLaunchKernelGGL(ln_kernel, dim3((rows + 7) / 8), dim3(256), 0, stream,
                       a, b2, g, bb, out, rows);
  };
  auto ufac = [&](int L) { int v = (int)(5.0 * ceil(log((double)L)));
                           return v < L ? v : L; };
  auto probAttn = [&](const float* q, const float* k, const float* v, float* ctx,
                      float* outUnmix, int L, int mask, int cums, unsigned seed) {
    int LK = L, u = ufac(L);
    hipLaunchKernelGGL(sample_m_kernel, grid1((long long)BATCH * NH * L), dim3(256),
                       0, stream, q, k, Mbuf, L, LK, u, seed);
    hipLaunchKernelGGL(topk_kernel, dim3(BATCH * NH), dim3(32), 0, stream,
                       Mbuf, idxTop, L, u);
    hipLaunchKernelGGL(scores_red_kernel, grid1((long long)BATCH * NH * u * LK),
                       dim3(256), 0, stream, q, k, idxTop, scRed, L, LK, u, mask);
    int rows = BATCH * NH * u;
    hipLaunchKernelGGL(softmax_rows, dim3((rows + 7) / 8), dim3(256), 0, stream,
                       scRed, rows, LK);
    if (cums)
      hipLaunchKernelGGL(ctx_cumsum_kernel, grid1(BATCH * NH * DH), dim3(256),
                         0, stream, v, ctx, L);
    else {
      hipLaunchKernelGGL(vmean_kernel, grid1(BATCH * NH * DH), dim3(256), 0, stream,
                         v, vmeanB, LK);
      hipLaunchKernelGGL(ctx_bcast_kernel, grid1((long long)BATCH * NH * L * DH),
                         dim3(256), 0, stream, vmeanB, ctx, L);
    }
    hipLaunchKernelGGL(ctx_scatter_kernel, grid1((long long)BATCH * NH * u * DH),
                       dim3(256), 0, stream, scRed, v, idxTop, ctx, L, LK, u);
    if (outUnmix)
      hipLaunchKernelGGL(unmix_kernel, grid1((long long)BATCH * L * DM), dim3(256),
                         0, stream, ctx, outUnmix, L);
  };

  // =========================== encoder ===========================
  int L = 2048;
  hipLaunchKernelGGL(embed_kernel, grid1((long long)BATCH * L * DM), dim3(256), 0,
                     stream, x_enc, x_mark_enc, eCw, eTw, eTb, ePe, bufA, L, 7);
  for (int i = 0; i < 3; ++i) {
    const EncL& e = enc[i];
    int M = BATCH * L;
    lin(bufA, e.a.wq, e.a.bq, bufQ, M, DM, DM, 0);
    lin(bufA, e.a.wk, e.a.bk, bufK, M, DM, DM, 0);
    lin(bufA, e.a.wv, e.a.bv, bufV, M, DM, DM, 0);
    probAttn(bufQ, bufK, bufV, ctxBuf, bufB, L, 0, 0, 1000u + i);
    lin(bufB, e.a.wo, e.a.bo, bufC, M, DM, DM, 0);
    ln(bufA, bufC, e.ln1g, e.ln1b, bufB, M);
    lin(bufB, e.w1, e.b1, big, M, DFF, DM, 1);          // GELU fused
    lin(big, e.w2, e.b2, bufC, M, DM, DFF, 0);
    ln(bufB, bufC, e.ln2g, e.ln2b, bufA, M);
    if (i < 2) {                                        // conv distillation
      int Lp4 = L + 4, Lc = L + 2;
      hipLaunchKernelGGL(wrappad_kernel, grid1((long long)BATCH * Lp4 * DM),
                         dim3(256), 0, stream, bufA, convXp, L, 2);
      for (int t = 0; t < 3; ++t)                       // 3 accumulating tap-GEMMs
        launch_gemm(stream, convXp + t * DM, cv[i].w + t,
                    (t == 0) ? cv[i].b : nullptr, convY, Lc, DM, DM,
                    DM, DM * 3, DM, 1.f, 0, 1, (t > 0), BATCH, 1,
                    (long long)Lp4 * DM, 0, 0, 0, (long long)Lc * DM, 0, 3);
      hipLaunchKernelGGL(bnelu_kernel, grid1((long long)BATCH * Lc * DM), dim3(256),
                         0, stream, convY, cv[i].bnm, cv[i].bnv, cv[i].bng,
                         cv[i].bnb, (long long)BATCH * Lc * DM);
      int Lp = (Lc - 1) / 2 + 1;
      hipLaunchKernelGGL(maxpool3_kernel, grid1((long long)BATCH * Lp * DM),
                         dim3(256), 0, stream, convY, bufA, Lc, Lp);
      L = Lp;
    }
  }
  ln(bufA, nullptr, encNg, encNb, encOut, BATCH * L);
  const int S = L;

  // =========================== decoder ===========================
  int Ld = 1024;
  hipLaunchKernelGGL(embed_kernel, grid1((long long)BATCH * Ld * DM), dim3(256), 0,
                     stream, x_dec, x_mark_dec, dCw, dTw, dTb, dPe, bufA, Ld, 7);
  for (int i = 0; i < 2; ++i) {
    const DecL& d = dec[i];
    int M = BATCH * Ld;
    lin(bufA, d.self.wq, d.self.bq, bufQ, M, DM, DM, 0);
    lin(bufA, d.self.wk, d.self.bk, bufK, M, DM, DM, 0);
    lin(bufA, d.self.wv, d.self.bv, bufV, M, DM, DM, 0);
    probAttn(bufQ, bufK, bufV, ctxBuf, nullptr, Ld, 1, 1, 2000u + i);
    lin(ctxBuf, d.self.wo, d.self.bo, bufC, M, DM, DM, 0);  // mix: flat reinterpret
    ln(bufC, bufC, d.ln1g, d.ln1b, bufB, M);                // LN(sa + sa)
    lin(bufB, d.cross.wq, d.cross.bq, bufQ, M, DM, DM, 0);
    lin(encOut, d.cross.wk, d.cross.bk, bufK, BATCH * S, DM, DM, 0);
    lin(encOut, d.cross.wv, d.cross.bv, bufV, BATCH * S, DM, DM, 0);
    // scores[b,h,Lq,S] = Q.K^T / 8
    launch_gemm(stream, bufQ, bufK, nullptr, big, Ld, S, DH, DM, DM, S,
                0.125f, 0, 1, 0, BATCH * NH, NH,
                (long long)Ld * DM, DH, (long long)S * DM, DH,
                (long long)NH * Ld * S, (long long)Ld * S, 1);
    { int rows = BATCH * NH * Ld;
      hipLaunchKernelGGL(softmax_rows, dim3((rows + 7) / 8), dim3(256), 0, stream,
                         big, rows, S); }
    // ctx[b,l,h*64+d] = probs @ V   (B is K-major)
    launch_gemm(stream, big, bufV, nullptr, bufA, Ld, DH, S, S, DM, DM,
                1.f, 0, 0, 0, BATCH * NH, NH,
                (long long)NH * Ld * S, (long long)Ld * S,
                (long long)S * DM, DH, (long long)Ld * DM, DH, 1);
    lin(bufA, d.cross.wo, d.cross.bo, bufQ, M, DM, DM, 0);
    ln(bufQ, bufQ, d.ln2g, d.ln2b, bufB, M);                // LN(ca + ca)
    lin(bufB, d.w1, d.b1, big, M, DFF, DM, 1);
    lin(big, d.w2, d.b2, bufC, M, DM, DFF, 0);
    ln(bufB, bufC, d.ln3g, d.ln3b, bufA, M);
  }
  ln(bufA, nullptr, decNg, decNb, bufB, BATCH * Ld);
  lin(bufB, projW, projB, projY, BATCH * Ld, 7, DM, 0);
  hipLaunchKernelGGL(slice_out_kernel, grid1((long long)BATCH * 512 * 7), dim3(256),
                     0, stream, projY, (float*)d_out, Ld, 512, 7);
}
